// GraphNeuralNetwork_61984968015976
// MI455X (gfx1250) — matrix operand order repaired and checked by reading, not compiled, not run on hardware
//
#include <hip/hip_runtime.h>
#include <hip/hip_bf16.h>
#include <stdint.h>

// ---------------------------------------------------------------------------
// GraphNeuralNetwork on MI455X (gfx1250, wave32, WMMA)
//
// Each sparse layer (64 random parents per output node) is densified into a
// dense weight matrix stored transposed (WT[j][p]) and split into bf16 hi/lo
// planes. The layer then becomes a dense GEMM
//     out[b,j] = sum_p act[b,p] * WT[j,p]
// executed with V_WMMA_F32_16X16X32_BF16 using bf16x3 emulation
// (hi*hi + hi*lo + lo*hi, fp32 accumulate) for ~fp32 accuracy.
// All tensors (16MB weights + 8MB activations) are L2-resident (192MB L2).
//
// GEMM tiling: block = 256 threads = 8 waves; block tile 64(M) x 128(N);
// wave tile 32x32 (2x2 register blocking -> 12 WMMAs per 32-K chunk per
// wave, hi fragments feed 4 WMMAs, lo fragments 2). Staging uses the CDNA5
// async global->LDS path (GLOBAL_LOAD_ASYNC_TO_LDS_B128 + s_wait_asynccnt)
// when the toolchain exposes it, else global_load+ds_store.
// ---------------------------------------------------------------------------

typedef __bf16 bf16;
typedef __attribute__((ext_vector_type(16))) __bf16 v16bf;
typedef __attribute__((ext_vector_type(8)))  float  v8f;
typedef __attribute__((ext_vector_type(4)))  int    v4i;

union FragBF { v16bf v; uint32_t u[8]; };

#if __has_builtin(__builtin_amdgcn_global_load_async_to_lds_b128)
#define USE_ASYNC_LDS 1
#else
#define USE_ASYNC_LDS 0
#endif

// copy 16B global -> LDS for this lane
static __device__ __forceinline__ void stage16(void* lds, const void* glb) {
#if USE_ASYNC_LDS
  __builtin_amdgcn_global_load_async_to_lds_b128(
      (__attribute__((address_space(1))) v4i*)(uintptr_t)glb,
      (__attribute__((address_space(3))) v4i*)(uint32_t)(uintptr_t)lds,
      0, 0);
#else
  *(uint4*)lds = *(const uint4*)glb;
#endif
}

static __device__ __forceinline__ void stage_wait() {
#if USE_ASYNC_LDS
#if __has_builtin(__builtin_amdgcn_s_wait_asynccnt)
  __builtin_amdgcn_s_wait_asynccnt(0);
#else
  asm volatile("s_wait_asynccnt 0" ::: "memory");
#endif
#endif
}

// --- fp32 -> bf16 hi/lo split of the network input -------------------------
__global__ __launch_bounds__(256) void k_split(const float* __restrict__ x,
                                               bf16* __restrict__ hi,
                                               bf16* __restrict__ lo, int n) {
  int i = blockIdx.x * 256 + threadIdx.x;
  if (i < n) {
    float v = x[i];
    bf16 h = (bf16)v;
    hi[i] = h;
    lo[i] = (bf16)(v - (float)h);
  }
}

// --- densify one layer: WT[j][p] = sum of w over edges (src=p -> dst=j) ----
// One block per output node j. The 64 edges are accumulated serially by a
// single thread into an LDS fp32 row (deterministic even with duplicate
// src indices; no atomics), then the row is split to bf16 hi/lo.
__global__ __launch_bounds__(256) void k_densify(const float* __restrict__ w,
                                                 const int* __restrict__ src,
                                                 int n_prev,
                                                 bf16* __restrict__ wt_hi,
                                                 bf16* __restrict__ wt_lo) {
  __shared__ float row[2048];
  const int j = blockIdx.x;
  for (int p = threadIdx.x; p < n_prev; p += 256) row[p] = 0.0f;
  __syncthreads();
  if (threadIdx.x == 0) {
    const float* wj = w + (size_t)j * 64;
    const int*   sj = src + (size_t)j * 64;
    for (int k = 0; k < 64; ++k) row[sj[k]] += wj[k];
  }
  __syncthreads();
  bf16* oh = wt_hi + (size_t)j * n_prev;
  bf16* ol = wt_lo + (size_t)j * n_prev;
  for (int p = threadIdx.x; p < n_prev; p += 256) {
    float v = row[p];
    bf16 h = (bf16)v;
    oh[p] = h;
    ol[p] = (bf16)(v - (float)h);
  }
}

// --- dense GEMM with WMMA bf16 (fp32 accum), bias + ReLU + re-split fused --
__global__ __launch_bounds__(256) void k_gemm(
    const bf16* __restrict__ a_hi, const bf16* __restrict__ a_lo,
    const bf16* __restrict__ wt_hi, const bf16* __restrict__ wt_lo,
    const float* __restrict__ bias,
    bf16* __restrict__ out_hi, bf16* __restrict__ out_lo,
    float* __restrict__ out_f32,
    int N, int K, int do_relu) {
  __shared__ __align__(16) bf16 AsH[64][32];
  __shared__ __align__(16) bf16 AsL[64][32];
  __shared__ __align__(16) bf16 BsH[128][32];
  __shared__ __align__(16) bf16 BsL[128][32];

  const int t    = threadIdx.x;
  const int lane = t & 31;          // wave32
  const int wave = t >> 5;          // 8 waves
  const int m0 = blockIdx.y * 64;
  const int j0 = blockIdx.x * 128;
  const int mt = wave >> 2;         // 0..1  -> 32-row wave tile
  const int nt = wave & 3;          // 0..3  -> 32-col wave tile

  // cooperative LDS fill indices (16B = 8 bf16 per thread per buffer)
  const int lr = t >> 2, lk = (t & 3) * 8;

  // fragment source indices (ISA 7.12.2 wave32 layouts)
  const int l15 = lane & 15;
  const int akb = (lane >> 4) * 8;     // A K base (0 or 8)
  const int bkb = (lane >> 4) * 8;     // B K base (0 or 8)

  v8f acc[2][2] = {{{}, {}}, {{}, {}}};

  for (int k0 = 0; k0 < K; k0 += 32) {
    // stage 64x32 A (hi/lo) and 128x32 B (hi/lo) chunks into LDS
    stage16(&AsH[lr][lk],      &a_hi[(size_t)(m0 + lr) * K + k0 + lk]);
    stage16(&AsL[lr][lk],      &a_lo[(size_t)(m0 + lr) * K + k0 + lk]);
    stage16(&BsH[lr][lk],      &wt_hi[(size_t)(j0 + lr) * K + k0 + lk]);
    stage16(&BsL[lr][lk],      &wt_lo[(size_t)(j0 + lr) * K + k0 + lk]);
    stage16(&BsH[64 + lr][lk], &wt_hi[(size_t)(j0 + 64 + lr) * K + k0 + lk]);
    stage16(&BsL[64 + lr][lk], &wt_lo[(size_t)(j0 + 64 + lr) * K + k0 + lk]);
    if (k0 + 32 < K) {  // keep next chunk warm in near caches
      __builtin_prefetch(&wt_hi[(size_t)(j0 + lr) * K + k0 + 32 + lk], 0, 3);
      __builtin_prefetch(&wt_hi[(size_t)(j0 + 64 + lr) * K + k0 + 32 + lk], 0, 3);
      __builtin_prefetch(&a_hi[(size_t)(m0 + lr) * K + k0 + 32 + lk], 0, 3);
    }
    stage_wait();
    __syncthreads();

    // build fragments (hi used by 4 WMMAs, lo by 2)
    FragBF aH[2], aL[2], bH[2], bL[2];
#pragma unroll
    for (int mi = 0; mi < 2; ++mi) {
      // A: lane<16 -> K {0..7,16..23}; lane>=16 -> K {8..15,24..31}
      const int rr = mt * 32 + mi * 16 + l15;
      const uint32_t* ph = (const uint32_t*)&AsH[rr][0];
      const uint32_t* pl = (const uint32_t*)&AsL[rr][0];
#pragma unroll
      for (int d = 0; d < 4; ++d) {
        aH[mi].u[d]     = ph[(akb >> 1) + d];
        aH[mi].u[4 + d] = ph[((akb + 16) >> 1) + d];
        aL[mi].u[d]     = pl[(akb >> 1) + d];
        aL[mi].u[4 + d] = pl[((akb + 16) >> 1) + d];
      }
    }
#pragma unroll
    for (int ni = 0; ni < 2; ++ni) {
      // B: lane<16 -> K {0..7,16..23}; lane>=16 -> K {8..15,24..31}
      const int cc = nt * 32 + ni * 16 + l15;
      const uint32_t* qh = (const uint32_t*)&BsH[cc][0];
      const uint32_t* ql = (const uint32_t*)&BsL[cc][0];
#pragma unroll
      for (int d = 0; d < 4; ++d) {
        bH[ni].u[d]     = qh[(bkb >> 1) + d];
        bH[ni].u[4 + d] = qh[((bkb + 16) >> 1) + d];
        bL[ni].u[d]     = ql[(bkb >> 1) + d];
        bL[ni].u[4 + d] = ql[((bkb + 16) >> 1) + d];
      }
    }

    // bf16x3: hi*hi + hi*lo + lo*hi, fp32 accumulation (12 WMMAs / chunk)
#pragma unroll
    for (int mi = 0; mi < 2; ++mi)
#pragma unroll
      for (int ni = 0; ni < 2; ++ni) {
        acc[mi][ni] = __builtin_amdgcn_wmma_f32_16x16x32_bf16(
            false, aH[mi].v, false, bH[ni].v, (short)0, acc[mi][ni], false, false);
        acc[mi][ni] = __builtin_amdgcn_wmma_f32_16x16x32_bf16(
            false, aH[mi].v, false, bL[ni].v, (short)0, acc[mi][ni], false, false);
        acc[mi][ni] = __builtin_amdgcn_wmma_f32_16x16x32_bf16(
            false, aL[mi].v, false, bH[ni].v, (short)0, acc[mi][ni], false, false);
      }
    __syncthreads();
  }

  // epilogue: bias + optional ReLU; fp32 out (last layer) or bf16 hi/lo
#pragma unroll
  for (int ni = 0; ni < 2; ++ni) {
    const int col = j0 + nt * 32 + ni * 16 + l15;
    const float bcol = bias[col];
#pragma unroll
    for (int mi = 0; mi < 2; ++mi) {
#pragma unroll
      for (int r = 0; r < 8; ++r) {
        const int rowg = m0 + mt * 32 + mi * 16 + ((lane < 16) ? r : r + 8);
        float v = acc[mi][ni][r] + bcol;
        if (do_relu) v = v > 0.0f ? v : 0.0f;
        if (out_f32) out_f32[(size_t)rowg * N + col] = v;
        if (out_hi) {
          bf16 h = (bf16)v;
          out_hi[(size_t)rowg * N + col] = h;
          out_lo[(size_t)rowg * N + col] = (bf16)(v - (float)h);
        }
      }
    }
  }
}

// ---------------------------------------------------------------------------
extern "C" void kernel_launch(void* const* d_in, const int* in_sizes, int n_in,
                              void* d_out, int out_size, void* d_ws, size_t ws_size,
                              hipStream_t stream) {
  (void)in_sizes; (void)n_in; (void)out_size; (void)ws_size;
  static const int S[5] = {512, 2048, 2048, 2048, 512};

  const float* x = (const float*)d_in[0];
  const float* w[4]; const float* b[4]; const int* src[4];
  for (int l = 0; l < 4; ++l) {
    w[l]   = (const float*)d_in[1 + 4 * l];
    b[l]   = (const float*)d_in[2 + 4 * l];
    src[l] = (const int*)  d_in[3 + 4 * l];
    // d_in[4 + 4*l] (dst) is implicit: repeat(arange(n_curr), 64)
  }

  // workspace layout (24 MB total)
  char* ws = (char*)d_ws;
  bf16* wtH = (bf16*)ws; ws += (size_t)2048 * 2048 * sizeof(bf16);
  bf16* wtL = (bf16*)ws; ws += (size_t)2048 * 2048 * sizeof(bf16);
  bf16* act[4];
  for (int i = 0; i < 4; ++i) { act[i] = (bf16*)ws; ws += (size_t)512 * 2048 * sizeof(bf16); }
  bf16 *curH = act[0], *curL = act[1], *nxtH = act[2], *nxtL = act[3];

  const int n0 = 512 * 512;
  k_split<<<(n0 + 255) / 256, 256, 0, stream>>>(x, curH, curL, n0);

  for (int l = 0; l < 4; ++l) {
    const int n_prev = S[l], n_curr = S[l + 1];
    const bool last = (l == 3);
    k_densify<<<n_curr, 256, 0, stream>>>(w[l], src[l], n_prev, wtH, wtL);
    k_gemm<<<dim3(n_curr / 128, 512 / 64), 256, 0, stream>>>(
        curH, curL, wtH, wtL, b[l],
        last ? nullptr : nxtH, last ? nullptr : nxtL,
        last ? (float*)d_out : nullptr,
        n_curr, n_prev, last ? 0 : 1);
    bf16* tH = curH; curH = nxtH; nxtH = tH;
    bf16* tL = curL; curL = nxtL; nxtL = tL;
  }
}
